// FuncTRFsGen_27135603376407
// MI455X (gfx1250) — matrix operand order repaired and checked by reading, not compiled, not run on hardware
//
#include <hip/hip_runtime.h>
#include <math.h>

typedef __attribute__((ext_vector_type(16))) _Float16 v16h;
typedef __attribute__((ext_vector_type(8)))  float    v8f;

namespace {
constexpr int kBatch     = 8;
constexpr int kSeq       = 4096;
constexpr int kWin       = 65;
constexpr int kOut       = 64;
constexpr int kBasis     = 21;   // 1 DC + 20 sin/cos
constexpr int kSTiles    = kSeq / 16;  // 256
constexpr int kWChunks   = 5;          // 5 * 13 = 65 windows
constexpr int kWPerChunk = 13;
}

// One wave per (batch, 16-seq tile, 13-window chunk).
// Per window: 4x v_wmma_f32_16x16x32_f16 computes the 64(out) x 16(seq)
// Fourier-basis contraction (K = 20 basis fns padded to 32), DC term folded
// into the C accumulator, then gain-scale + streaming NT stores.
__global__ __launch_bounds__(32)
void trf_wmma_kernel(const float* __restrict__ x,
                     const float* __restrict__ conv_w,
                     const float* __restrict__ conv_b,
                     const float* __restrict__ coefs,
                     float* __restrict__ out)
{
    const float kTwoPiOverT = 0.0805536582f;  // 2*pi / 78
    const float kScale      = 0.1601281538f;  // 1/sqrt(78/2)
    const float kConst0     = 0.1132277034f;  // 1/(sqrt(2)*sqrt(39))
    const float kLimit      = 7.0f;

    const int bid  = blockIdx.x;
    const int wc   = bid % kWChunks;
    const int st   = (bid / kWChunks) % kSTiles;
    const int b    = bid / (kWChunks * kSTiles);

    const int lane = threadIdx.x;   // wave32
    const int half = lane >> 4;     // 0: lanes 0-15, 1: lanes 16-31
    const int l    = lane & 15;

    // ---- A operand: coefs[o, 1+K] * scale, 4 o-tiles, 16x32 f16 each ----
    // Layout (16-bit A 16x32): lanes 0-15 M=l hold K 0-15 (elems 0-15);
    // lanes 16-31 hold K 16-31 (matches B's contiguous K order).
    v16h A[4];
    #pragma unroll
    for (int ot = 0; ot < 4; ++ot) {
        const int o = ot * 16 + l;
        #pragma unroll
        for (int e = 0; e < 16; ++e) {
            int K;
            if (half == 0) K = e;
            else           K = 16 + e;
            const float v = (K < 20) ? coefs[o * kBasis + 1 + K] * kScale : 0.0f;
            A[ot][e] = (_Float16)v;
        }
    }

    // ---- C operand: DC term const0*coefs[o,0], broadcast over columns ----
    // C/D layout: VGPR r -> M = r (lanes 0-15) / r+8 (lanes 16-31), N = lane%16.
    v8f C[4];
    #pragma unroll
    for (int ot = 0; ot < 4; ++ot) {
        #pragma unroll
        for (int r = 0; r < 8; ++r) {
            const int o = ot * 16 + half * 8 + r;
            C[ot][r] = kConst0 * coefs[o * kBasis];
        }
    }

    // ---- Warp params from causal conv (k=2, left pad 1) at s = st*16 + l ----
    const int   s   = st * 16 + l;
    const float xs  = x[b * kSeq + s];
    const float xm1 = (s > 0) ? x[b * kSeq + s - 1] : 0.0f;
    const float p0  = fmaf(xm1, conv_w[0], fmaf(xs, conv_w[1], conv_b[0]));
    const float p1  = fmaf(xm1, conv_w[2], fmaf(xs, conv_w[3], conv_b[1]));
    const float p2  = fmaf(xm1, conv_w[4], fmaf(xs, conv_w[5], conv_b[2]));
    const float gain = fabsf(p0);
    const float bsh  = fminf(fmaxf(p1, -kLimit), kLimit);
    const float cdil = fminf(fmaxf(1.0f + p2, 0.5f), 1.4f);

    const int w0 = wc * kWPerChunk;
    for (int w = w0; w < w0 + kWPerChunk; ++w) {
        const float t     = cdil * ((float)(w + 7) - bsh);
        const float theta = kTwoPiOverT * t;

        // one sincos + angle-addition recurrence for harmonics 2..10
        float hs[10], hc[10];
        __sincosf(theta, &hs[0], &hc[0]);
        #pragma unroll
        for (int m = 1; m < 10; ++m) {
            hs[m] = fmaf(hs[m - 1], hc[0],  hc[m - 1] * hs[0]);
            hc[m] = fmaf(hc[m - 1], hc[0], -(hs[m - 1] * hs[0]));
        }

        // ---- B operand: 32(K=basis) x 16(N=seq) f16 ----
        // lanes 0-15 hold K 0-15 (harmonics 1-8), lanes 16-31 hold K 16-31
        // (harmonics 9-10 valid, rest zero-padded).
        v16h B;
        if (half == 0) {
            #pragma unroll
            for (int m = 0; m < 8; ++m) {
                B[2 * m]     = (_Float16)hs[m];
                B[2 * m + 1] = (_Float16)hc[m];
            }
        } else {
            B[0] = (_Float16)hs[8];
            B[1] = (_Float16)hc[8];
            B[2] = (_Float16)hs[9];
            B[3] = (_Float16)hc[9];
            #pragma unroll
            for (int e = 4; e < 16; ++e) B[e] = (_Float16)0.0f;
        }

        #pragma unroll
        for (int ot = 0; ot < 4; ++ot) {
            v8f D = __builtin_amdgcn_wmma_f32_16x16x32_f16(
                false, A[ot], false, B, (short)0, C[ot], false, false);
            #pragma unroll
            for (int r = 0; r < 8; ++r) {
                const int o = ot * 16 + half * 8 + r;
                const long long idx =
                    (((long long)(b * kOut + o)) * kWin + w) * kSeq + s;
                __builtin_nontemporal_store(gain * D[r], out + idx);
            }
        }
    }
}

extern "C" void kernel_launch(void* const* d_in, const int* in_sizes, int n_in,
                              void* d_out, int out_size, void* d_ws, size_t ws_size,
                              hipStream_t stream) {
    (void)in_sizes; (void)n_in; (void)out_size; (void)d_ws; (void)ws_size;
    const float* x      = (const float*)d_in[0];
    const float* conv_w = (const float*)d_in[1];
    const float* conv_b = (const float*)d_in[2];
    const float* coefs  = (const float*)d_in[3];
    float* out          = (float*)d_out;

    const int nBlocks = kBatch * kSTiles * kWChunks;  // 8*256*5 = 10240 waves
    trf_wmma_kernel<<<nBlocks, 32, 0, stream>>>(x, conv_w, conv_b, coefs, out);
}
